// Attention_7301444403681
// MI455X (gfx1250) — compile-verified, matrix operand order checked by reading, not run on hardware
//
#include <hip/hip_runtime.h>
#include <hip/hip_bf16.h>
#include <math.h>

// ---------------------------------------------------------------------------
// Fused causal attention, MI455X (gfx1250), wave32 + WMMA bf16.
//   B=4, T=2048, C=1024.  qkv projection (V stored transposed) -> flash attn.
// Workspace: qk [B][T][2C] bf16 (32 MB) + v^T [B][C][T] bf16 (16 MB); both
// L2-resident (192 MB), so attention streams K rows / V^T rows from L2.
// ---------------------------------------------------------------------------

#define BATCH   4
#define SEQ     2048
#define EMB     1024
#define EMB2    2048
#define EMB3    3072
#define BT      (BATCH * SEQ)          // 8192 rows for the qkv GEMM
#define SCALE_QK 0.03125f              // 1/sqrt(1024)

typedef __bf16 bf16;
typedef __attribute__((ext_vector_type(16))) __bf16 v16bf;
typedef __attribute__((ext_vector_type(8)))  __bf16 v8bf;
typedef __attribute__((ext_vector_type(8)))  float  v8f;

// 16x32 bf16 A/B fragment (8 VGPRs): two contiguous 8-element (16B) chunks.
union Frag {
    v16bf v;
    v8bf  h[2];
};

__device__ inline v8f wmma_bf16(const Frag& a, const Frag& b, v8f c) {
    // (neg_a, A, neg_b, B, c_mod, C, reuse_a, reuse_b)
    return __builtin_amdgcn_wmma_f32_16x16x32_bf16(
        false, a.v, false, b.v, (short)0, c, false, false);
}

// Load an A/B fragment from a row-major [16 rows][ld>=32] bf16 tile.
// Per ISA 7.12.2 (16-bit 16x32 operand):
//   lanes 0-15 : row = lane,    chunk0 k=0..7,  chunk1 k=16..23
//   lanes 16-31: row = lane-16, chunk0 k=8..15, chunk1 k=24..31
__device__ inline Frag load_frag(const bf16* __restrict__ base, int ld, int lane) {
    Frag f;
    const bf16* row = base + (size_t)(lane & 15) * ld;
    int kof = (lane >> 4) << 3;                  // 0 or 8
    f.h[0] = *(const v8bf*)(row + kof);
    f.h[1] = *(const v8bf*)(row + 16 + kof);
    return f;
}

// ---------------------------------------------------------------------------
// Kernel 1: qkv = x @ W,  M=8192 K=1024 N=3072, fp32 in -> bf16 out.
// Workgroup = 4 waves, 64x128 tile; wave w owns N-slab [w*32, w*32+32)
// (4 M-tiles x 2 N-tiles = 8 WMMA per K-step, A-frags reused across B).
// Epilogue (block-uniform branch): cols [0,2C) -> qk row-major;
// cols [2C,3C) -> v^T [b][dim][t] with packed 16B stores.
// ---------------------------------------------------------------------------
__global__ __launch_bounds__(128) void qkv_gemm_kernel(
    const float* __restrict__ x, const float* __restrict__ w,
    bf16* __restrict__ qk, bf16* __restrict__ vt)
{
    __shared__ bf16 xs[64 * 32];      // x tile  [m][k], bf16
    __shared__ bf16 wts[128 * 32];    // W tile transposed [n][k], bf16

    const int wid  = threadIdx.x >> 5;
    const int lane = threadIdx.x & 31;
    const int mbase = blockIdx.x * 64;
    const int nbase = blockIdx.y * 128;

    v8f acc[4][2] = {};               // [M-tile][N-tile] 16x16 f32

    for (int k0 = 0; k0 < EMB; k0 += 32) {
        __syncthreads();
        // stage x tile (64x32): 16 contiguous floats / thread, cvt to bf16
        {
            int m  = threadIdx.x >> 1;
            int kk = (threadIdx.x & 1) * 16;
            const float* src = x + (size_t)(mbase + m) * EMB + k0 + kk;
            bf16* dst = xs + m * 32 + kk;
            #pragma unroll
            for (int i = 0; i < 16; ++i) dst[i] = (bf16)src[i];
        }
        // stage W^T tile (128x32): thread owns column n = tid (coalesced
        // across lanes for each k), writes contiguous wts[n][0..31]
        {
            int n = threadIdx.x;
            const float* src = w + (size_t)k0 * EMB3 + nbase + n;
            bf16* dst = wts + n * 32;
            #pragma unroll
            for (int k = 0; k < 32; ++k) dst[k] = (bf16)src[(size_t)k * EMB3];
        }
        __syncthreads();

        Frag bfr[2];
        #pragma unroll
        for (int nt = 0; nt < 2; ++nt)
            bfr[nt] = load_frag(wts + (size_t)(wid * 32 + nt * 16) * 32, 32, lane);
        #pragma unroll
        for (int mt = 0; mt < 4; ++mt) {
            Frag af = load_frag(xs + (size_t)(mt * 16) * 32, 32, lane);
            #pragma unroll
            for (int nt = 0; nt < 2; ++nt)
                acc[mt][nt] = wmma_bf16(af, bfr[nt], acc[mt][nt]);
        }
    }

    // D layout: VGPR r -> row r (lanes 0-15) / 8+r (lanes 16-31), col lane&15
    const int rofs = (lane >> 4) * 8;
    const int lc   = lane & 15;
    if (nbase < EMB2) {
        // ---- q,k region: row-major [row][col], scalar b16 stores ----
        #pragma unroll
        for (int nt = 0; nt < 2; ++nt) {
            const int col = nbase + wid * 32 + nt * 16 + lc;
            #pragma unroll
            for (int mt = 0; mt < 4; ++mt) {
                bf16* dst = qk + (size_t)(mbase + mt * 16 + rofs) * EMB2 + col;
                #pragma unroll
                for (int r = 0; r < 8; ++r)
                    dst[(size_t)r * EMB2] = (bf16)acc[mt][nt][r];
            }
        }
    } else {
        // ---- v region: [b][dim][t]; 8 rows/lane are contiguous t -> b128 ----
        const int b = mbase >> 11;                 // 64-row tiles never span b
        const int tbase = mbase & (SEQ - 1);
        #pragma unroll
        for (int nt = 0; nt < 2; ++nt) {
            const int d = nbase - EMB2 + wid * 32 + nt * 16 + lc;
            #pragma unroll
            for (int mt = 0; mt < 4; ++mt) {
                v8bf pk;
                #pragma unroll
                for (int r = 0; r < 8; ++r) pk[r] = (bf16)acc[mt][nt][r];
                *(v8bf*)(vt + ((size_t)b * EMB + d) * SEQ + tbase + mt * 16 + rofs) = pk;
            }
        }
    }
}

// ---------------------------------------------------------------------------
// Kernel 2: flash attention, 4 waves / 16 Q rows per workgroup.
// Wave w: scores reduce Q.K^T over dims [w*256,+256); output owns the same
// dim slice. K rows and V^T rows are read straight from global (L2-hot) as
// WMMA B-fragments -- no LDS transposes. LDS only for the 16x32 f32 score
// reduction, bf16 P tile, and online-softmax state.
// ---------------------------------------------------------------------------
__global__ __launch_bounds__(128) void flash_attn_kernel(
    const bf16* __restrict__ qk, const bf16* __restrict__ vt,
    float* __restrict__ out)
{
    __shared__ float s_lds[16 * 32];        // score tile (cross-wave reduce)
    __shared__ bf16  p_lds[16 * 32];        // softmaxed probs
    __shared__ float row_m[16], row_l[16], row_scale[16];

    const int wid  = threadIdx.x >> 5;
    const int lane = threadIdx.x & 31;
    const int b        = blockIdx.y;
    const int row_base = blockIdx.x * 16;

    const bf16* Qp = qk + (size_t)b * SEQ * EMB2;          // q: cols [0,C)
    const bf16* Kp = Qp + EMB;                             // k: cols [C,2C)
    const bf16* Vt = vt + (size_t)b * EMB * SEQ;           // v^T [dim][t]

    const int m   = lane & 15;
    const int kof = (lane >> 4) << 3;

    // Q fragments for this wave's 256-dim slice (kept in registers)
    Frag qf[8];
    #pragma unroll
    for (int s = 0; s < 8; ++s) {
        const bf16* qrow = Qp + (size_t)(row_base + m) * EMB2 + wid * 256 + s * 32;
        qf[s].h[0] = *(const v8bf*)(qrow + kof);
        qf[s].h[1] = *(const v8bf*)(qrow + 16 + kof);
    }

    v8f acc[16] = {};                       // output dims wid*256 .. +256

    if (threadIdx.x < 16) { row_m[threadIdx.x] = -INFINITY; row_l[threadIdx.x] = 0.f; }

    const int kv_blocks = (row_base + 16 + 31) / 32;       // causal bound
    for (int kb = 0; kb < kv_blocks; ++kb) {
        const int kv0 = kb * 32;
        __syncthreads();
        for (int i = threadIdx.x; i < 16 * 32; i += 128) s_lds[i] = 0.f;
        __syncthreads();

        if (kb + 1 < kv_blocks)             // warm L2 for next K block
            __builtin_prefetch(Kp + (size_t)(kv0 + 32 + (threadIdx.x & 31)) * EMB2, 0, 1);

        // ---- partial scores S = Q . K^T over this wave's 256 dims ----
        v8f sacc[2] = {};
        #pragma unroll
        for (int nt = 0; nt < 2; ++nt) {
            #pragma unroll
            for (int s = 0; s < 8; ++s) {
                Frag bf;    // K row-major [seq][dim] == B^T layout directly
                const bf16* krow = Kp + (size_t)(kv0 + nt * 16 + m) * EMB2 + wid * 256 + s * 32;
                bf.h[0] = *(const v8bf*)(krow + kof);
                bf.h[1] = *(const v8bf*)(krow + 16 + kof);
                sacc[nt] = wmma_bf16(qf[s], bf, sacc[nt]);
            }
        }
        // ---- cross-wave reduce into LDS ----
        #pragma unroll
        for (int nt = 0; nt < 2; ++nt) {
            int colL = nt * 16 + (lane & 15);
            int rofsL = (lane >> 4) * 8;
            #pragma unroll
            for (int r = 0; r < 8; ++r)
                atomicAdd(&s_lds[(rofsL + r) * 32 + colL], sacc[nt][r]);
        }
        __syncthreads();

        // ---- online softmax update: one thread per Q row ----
        if (threadIdx.x < 16) {
            int r = threadIdx.x;
            int grow = row_base + r;
            float mx = row_m[r];
            float sv[32];
            #pragma unroll
            for (int c2 = 0; c2 < 32; ++c2) {
                int gc = kv0 + c2;
                float v = (gc <= grow) ? s_lds[r * 32 + c2] * SCALE_QK : -INFINITY;
                sv[c2] = v;
                mx = fmaxf(mx, v);
            }
            float rs = __expf(row_m[r] - mx);      // old contribution rescale
            float sum = 0.f;
            #pragma unroll
            for (int c2 = 0; c2 < 32; ++c2) {
                float p = __expf(sv[c2] - mx);
                sum += p;
                p_lds[r * 32 + c2] = (bf16)p;
            }
            row_scale[r] = rs;
            row_m[r] = mx;
            row_l[r] = row_l[r] * rs + sum;
        }
        __syncthreads();

        // ---- rescale accumulators ----
        float rs[8];
        #pragma unroll
        for (int r = 0; r < 8; ++r) rs[r] = row_scale[(lane >> 4) * 8 + r];
        #pragma unroll
        for (int t = 0; t < 16; ++t)
            #pragma unroll
            for (int r = 0; r < 8; ++r) acc[t][r] *= rs[r];

        // ---- O += P @ V : B-frags straight from global v^T [dim][t] ----
        Frag pf = load_frag(p_lds, 32, lane);
        #pragma unroll
        for (int nt = 0; nt < 16; ++nt) {
            Frag bf;
            const bf16* vrow = Vt + (size_t)(wid * 256 + nt * 16 + m) * SEQ + kv0;
            bf.h[0] = *(const v8bf*)(vrow + kof);
            bf.h[1] = *(const v8bf*)(vrow + 16 + kof);
            acc[nt] = wmma_bf16(pf, bf, acc[nt]);
        }
    }

    // ---- normalize by l and store ----
    __syncthreads();
    float invl[8];
    #pragma unroll
    for (int r = 0; r < 8; ++r) invl[r] = 1.0f / row_l[(lane >> 4) * 8 + r];
    const int rofs = (lane >> 4) * 8;
    #pragma unroll
    for (int nt = 0; nt < 16; ++nt) {
        int col = wid * 256 + nt * 16 + (lane & 15);
        #pragma unroll
        for (int r = 0; r < 8; ++r) {
            out[((size_t)b * SEQ + row_base + rofs + r) * EMB + col] =
                acc[nt][r] * invl[r];
        }
    }
}

// ---------------------------------------------------------------------------
extern "C" void kernel_launch(void* const* d_in, const int* in_sizes, int n_in,
                              void* d_out, int out_size, void* d_ws, size_t ws_size,
                              hipStream_t stream)
{
    const float* x = (const float*)d_in[0];       // [4, 2048, 1024] f32
    const float* w = (const float*)d_in[1];       // [1024, 3072]   f32
    float* out = (float*)d_out;                   // [4, 2048, 1024] f32
    bf16* qkbuf = (bf16*)d_ws;                    // [4][2048][2048] bf16, 32 MB
    bf16* vtbuf = qkbuf + (size_t)BT * EMB2;      // [4][1024][2048] bf16, 16 MB

    dim3 g1(BT / 64, EMB3 / 128);                 // 128 x 24 workgroups
    qkv_gemm_kernel<<<g1, 128, 0, stream>>>(x, w, qkbuf, vtbuf);

    dim3 g2(SEQ / 16, BATCH);                     // 128 x 4 workgroups
    flash_attn_kernel<<<g2, 128, 0, stream>>>(qkbuf, vtbuf, out);
}
